// ST_SAGCN_46059229283252
// MI455X (gfx1250) — matrix operand order repaired and checked by reading, not compiled, not run on hardware
//
#include <hip/hip_runtime.h>
#include <hip/hip_bf16.h>
#include <math.h>

// ---------------------------------------------------------------------------
// ST-SAGCN fused inference for gfx1250 (MI455X, wave32, WMMA 16x16x32 f16)
// One workgroup (256 thr = 8 waves) per batch sample; full 420x64 feature
// map lives in LDS as f16; all E x E GEMMs use v_wmma_f32_16x16x32_f16.
// ---------------------------------------------------------------------------

#define NB   1024
#define CCH  3
#define TT   30
#define VV   14
#define EE   64
#define HH   4
#define DD   16
#define NC   7
#define TOK  (TT * VV)       // 420
#define MT   27              // M tiles of 16 -> 432 padded rows
#define TOKP (MT * 16)       // 432

typedef __attribute__((ext_vector_type(16))) _Float16 v16h;
typedef __attribute__((ext_vector_type(8)))  _Float16 v8h;
typedef __attribute__((ext_vector_type(8)))  float    v8f;

struct Params {
    const float *x_joint, *x_vel;
    const float *je_w, *je_b, *ve_w, *ve_b;
    const float *joint_emb, *frame_emb;
    const float *proj_w, *proj_b;
    const float *gcn_w1, *gcn_b1, *gcn_w2, *gcn_b2;
    const float *sa_wq, *sa_bq, *sa_wk, *sa_bk, *sa_wv, *sa_bv, *sa_wo, *sa_bo;
    const float *ta_wq, *ta_bq, *ta_wk, *ta_bk, *ta_wv, *ta_bv, *ta_wo, *ta_bo;
    const float *fc_w, *fc_b;
    const float *terms;      // d_ws: jointTerm[14*64] then frameTerm[30*64]
    float       *out;        // [N, 7]
    int          n_batch;
};

// ---- precompute batch-invariant embedding terms into workspace -------------
__global__ void precompute_terms(const float* __restrict__ joint_emb,
                                 const float* __restrict__ frame_emb,
                                 const float* __restrict__ proj_w,
                                 float* __restrict__ terms) {
    const int tid = threadIdx.x;
    for (int i = tid; i < VV * EE; i += 256) {          // jointTerm
        const int v = i >> 6, e = i & 63;
        float a = 0.f;
        for (int k = 0; k < EE; ++k)
            a += joint_emb[v * EE + k] * proj_w[(EE + k) * EE + e];
        terms[i] = a;
    }
    for (int i = tid; i < TT * EE; i += 256) {          // frameTerm
        const int t = i >> 6, e = i & 63;
        float a = 0.f;
        for (int k = 0; k < EE; ++k)
            a += frame_emb[t * EE + k] * proj_w[(2 * EE + k) * EE + e];
        terms[VV * EE + i] = a;
    }
}

// ---- WMMA fragment loaders -------------------------------------------------
// A: 16x32 f16, rows striped across lanes per CDNA5 ISA 7.12.2
__device__ __forceinline__ v16h loadA(const _Float16* __restrict__ src,
                                      int m0, int k0, int lane) {
    const int r  = m0 + (lane & 15);
    const int kb = k0 + ((lane & 16) ? 16 : 0);
    v8h lo = *(const v8h*)(src + r * EE + kb);
    v8h hi = *(const v8h*)(src + r * EE + kb + 8);
    v16h a;
#pragma unroll
    for (int i = 0; i < 8; ++i) { a[i] = lo[i]; a[i + 8] = hi[i]; }
    return a;
}

// B: 32x16 f16 from transposed-weight LDS sWt[n][k] -> one 32B load per lane
__device__ __forceinline__ v16h loadB(const _Float16* __restrict__ sWt,
                                      int n0, int k0, int lane) {
    const int n  = n0 + (lane & 15);
    const int kb = k0 + ((lane & 16) ? 16 : 0);
    return *(const v16h*)(sWt + n * EE + kb);
}

// ---- generic fused GEMM: dst = [relu](src @ W + b) [+ resid] ---------------
__device__ __forceinline__ void gemmEE(const _Float16* __restrict__ src,
                                       _Float16* __restrict__ dst,
                                       const _Float16* __restrict__ sWt,
                                       const float* __restrict__ sBias,
                                       const _Float16* __restrict__ resid,
                                       bool relu, int lane, int wave) {
    v16h bf[4][2];
#pragma unroll
    for (int nt = 0; nt < 4; ++nt) {
        bf[nt][0] = loadB(sWt, nt * 16, 0,  lane);
        bf[nt][1] = loadB(sWt, nt * 16, 32, lane);
    }
    for (int mt = wave; mt < MT; mt += 8) {
        const int m0 = mt * 16;
        const v16h a0 = loadA(src, m0, 0,  lane);
        const v16h a1 = loadA(src, m0, 32, lane);
        const int rb = m0 + ((lane & 16) ? 8 : 0);
        const int cl = lane & 15;
#pragma unroll
        for (int nt = 0; nt < 4; ++nt) {
            v8f acc = {};
            acc = __builtin_amdgcn_wmma_f32_16x16x32_f16(
                false, a0, false, bf[nt][0], (short)0, acc, false, false);
            acc = __builtin_amdgcn_wmma_f32_16x16x32_f16(
                false, a1, false, bf[nt][1], (short)0, acc, false, false);
            const int col = nt * 16 + cl;
            const float bb = sBias[col];
#pragma unroll
            for (int r = 0; r < 8; ++r) {
                float o = acc[r] + bb;
                if (relu)  o = fmaxf(o, 0.f);
                if (resid) o += (float)resid[(rb + r) * EE + col];
                dst[(rb + r) * EE + col] = (_Float16)o;
            }
        }
    }
}

// ---- weight staging: global f32 W[k][n] -> LDS f16 transposed sWt[n][k] ----
__device__ __forceinline__ void stageW(const float* __restrict__ w,
                                       const float* __restrict__ b,
                                       _Float16* __restrict__ sWt,
                                       float* __restrict__ sBias, int tid) {
    for (int i = tid; i < EE * EE; i += 256) {
        const int k = i >> 6, n = i & 63;
        sWt[n * EE + k] = (_Float16)w[i];
    }
    if (tid < EE) sBias[tid] = b[tid];
}

// ---------------------------------------------------------------------------
__global__ __launch_bounds__(256, 1) void st_sagcn_fused(Params p) {
    __shared__ _Float16 sX[TOKP * EE];   // master feature map
    __shared__ _Float16 sQ[TOKP * EE];   // scratch / q / attn-out
    __shared__ _Float16 sK[TOKP * EE];   // scratch / k
    __shared__ _Float16 sV[TOKP * EE];   // scratch / v
    __shared__ _Float16 sWt[EE * EE];    // staged transposed weight (f16)
    __shared__ float    sBias[EE];
    __shared__ float    sPool[EE];

    const int tid  = threadIdx.x;
    const int lane = tid & 31;
    const int wave = tid >> 5;
    const int n    = blockIdx.x;

    // ---------------- stage 0: channel embed ------------------------------
    // alias free buffers: sQ <- raw inputs (f32), sK <- small weights/terms
    float* xin = (float*)sQ;             // [0,1260) x_joint, [1260,2520) x_vel
    float* wb  = (float*)sK;             // je_w|ve_w|je_b|ve_b|jointT|frameT
    {
        const float* xj = p.x_joint + (size_t)n * CCH * TOK;
        const float* xv = p.x_vel   + (size_t)n * CCH * TOK;
        for (int i = tid; i < CCH * TOK; i += 256) {
            xin[i]             = xj[i];
            xin[CCH * TOK + i] = xv[i];
        }
        for (int i = tid; i < CCH * EE; i += 256) {
            wb[i]            = p.je_w[i];
            wb[CCH * EE + i] = p.ve_w[i];
        }
        if (tid < EE) { wb[384 + tid] = p.je_b[tid]; wb[448 + tid] = p.ve_b[tid]; }
        for (int i = tid; i < VV * EE; i += 256) wb[512 + i]  = p.terms[i];
        for (int i = tid; i < TT * EE; i += 256) wb[1408 + i] = p.terms[VV * EE + i];
        stageW(p.proj_w, p.proj_b, sWt, sBias, tid);   // W1 = proj_w rows 0..63
    }
    __syncthreads();
    for (int i = tid; i < TOKP * EE; i += 256) {
        const int tok = i >> 6, e = i & 63;
        float a = 0.f;
        if (tok < TOK) {
            a = wb[384 + e] + wb[448 + e];
#pragma unroll
            for (int c = 0; c < CCH; ++c)
                a += xin[c * TOK + tok]            * wb[c * EE + e]
                   + xin[CCH * TOK + c * TOK + tok] * wb[CCH * EE + c * EE + e];
        }
        sX[i] = (_Float16)a;
    }
    __syncthreads();

    // ---------------- stage 1: proj GEMM + embedding terms -----------------
    gemmEE(sX, sQ, sWt, sBias, nullptr, false, lane, wave);
    __syncthreads();
    for (int i = tid; i < TOK * EE; i += 256) {
        const int tok = i >> 6, e = i & 63;
        const int v = tok % VV, t = tok / VV;
        sX[i] = (_Float16)((float)sQ[i] + wb[512 + v * EE + e] + wb[1408 + t * EE + e]);
    }
    __syncthreads();

    // ---------------- stage 2: cosine adjacency over joints ---------------
    float* rn = (float*)sV;              // 1/||row||  (sV free here)
    float* aw = (float*)sK;              // softmax weights [420][14] (sK free)
    for (int tok = tid; tok < TOK; tok += 256) {
        float s = 0.f;
        for (int e = 0; e < EE; ++e) {
            const float x = (float)sX[tok * EE + e];
            s += x * x;
        }
        rn[tok] = 1.f / fmaxf(sqrtf(s), 1e-12f);
    }
    __syncthreads();
    for (int u = tid; u < TOK; u += 256) {           // u = (t,i)
        const int t = u / VV;
        float sc[VV], mx = -1e30f;
#pragma unroll
        for (int j = 0; j < VV; ++j) {
            const int rj = t * VV + j;
            float s = 0.f;
            for (int e = 0; e < EE; ++e)
                s += (float)sX[u * EE + e] * (float)sX[rj * EE + e];
            s *= rn[u] * rn[rj];
            sc[j] = s; mx = fmaxf(mx, s);
        }
        float den = 0.f;
#pragma unroll
        for (int j = 0; j < VV; ++j) { sc[j] = expf(sc[j] - mx); den += sc[j]; }
        const float inv = 1.f / den;
#pragma unroll
        for (int j = 0; j < VV; ++j) aw[u * VV + j] = sc[j] * inv;
    }
    __syncthreads();
    for (int i = tid; i < TOK * EE; i += 256) {      // feat = adj @ feat -> sQ
        const int tok = i >> 6, e = i & 63, t = tok / VV;
        float a = 0.f;
#pragma unroll
        for (int j = 0; j < VV; ++j)
            a += aw[tok * VV + j] * (float)sX[(t * VV + j) * EE + e];
        sQ[i] = (_Float16)a;
    }
    __syncthreads();

    // ---------------- stage 3: GCN ----------------------------------------
    stageW(p.gcn_w1, p.gcn_b1, sWt, sBias, tid);  __syncthreads();
    gemmEE(sQ, sX, sWt, sBias, nullptr, true,  lane, wave);  __syncthreads();
    stageW(p.gcn_w2, p.gcn_b2, sWt, sBias, tid); __syncthreads();
    gemmEE(sX, sX, sWt, sBias, nullptr, false, lane, wave);  __syncthreads();

    // ---------------- stage 4: spatial MHSA (over V) + residual ------------
    stageW(p.sa_wq, p.sa_bq, sWt, sBias, tid); __syncthreads();
    gemmEE(sX, sQ, sWt, sBias, nullptr, false, lane, wave); __syncthreads();
    stageW(p.sa_wk, p.sa_bk, sWt, sBias, tid); __syncthreads();
    gemmEE(sX, sK, sWt, sBias, nullptr, false, lane, wave); __syncthreads();
    stageW(p.sa_wv, p.sa_bv, sWt, sBias, tid); __syncthreads();
    gemmEE(sX, sV, sWt, sBias, nullptr, false, lane, wave); __syncthreads();
    for (int u = tid; u < TT * HH * VV; u += 256) {
        const int i = u % VV, h = (u / VV) & 3, t = u / (VV * HH);
        const int c0 = h * DD, base = t * VV, rowi = base + i;
        float qv[DD];
#pragma unroll
        for (int d = 0; d < DD; ++d) qv[d] = (float)sQ[rowi * EE + c0 + d];
        float sc[VV], mx = -1e30f;
#pragma unroll
        for (int j = 0; j < VV; ++j) {
            const int rj = base + j;
            float s = 0.f;
#pragma unroll
            for (int d = 0; d < DD; ++d) s += qv[d] * (float)sK[rj * EE + c0 + d];
            s *= 0.25f; sc[j] = s; mx = fmaxf(mx, s);
        }
        float den = 0.f;
#pragma unroll
        for (int j = 0; j < VV; ++j) { sc[j] = expf(sc[j] - mx); den += sc[j]; }
        const float inv = 1.f / den;
        float ov[DD];
#pragma unroll
        for (int d = 0; d < DD; ++d) ov[d] = 0.f;
#pragma unroll
        for (int j = 0; j < VV; ++j) {
            const float pw = sc[j] * inv; const int rj = base + j;
#pragma unroll
            for (int d = 0; d < DD; ++d) ov[d] += pw * (float)sV[rj * EE + c0 + d];
        }
#pragma unroll
        for (int d = 0; d < DD; ++d) sQ[rowi * EE + c0 + d] = (_Float16)ov[d];
    }
    __syncthreads();
    stageW(p.sa_wo, p.sa_bo, sWt, sBias, tid); __syncthreads();
    gemmEE(sQ, sX, sWt, sBias, sX, false, lane, wave); __syncthreads();

    // ---------------- stage 5: temporal MHSA (over T) + residual -----------
    stageW(p.ta_wq, p.ta_bq, sWt, sBias, tid); __syncthreads();
    gemmEE(sX, sQ, sWt, sBias, nullptr, false, lane, wave); __syncthreads();
    stageW(p.ta_wk, p.ta_bk, sWt, sBias, tid); __syncthreads();
    gemmEE(sX, sK, sWt, sBias, nullptr, false, lane, wave); __syncthreads();
    stageW(p.ta_wv, p.ta_bv, sWt, sBias, tid); __syncthreads();
    gemmEE(sX, sV, sWt, sBias, nullptr, false, lane, wave); __syncthreads();
    for (int u = tid; u < VV * HH * TT; u += 256) {
        const int i = u % TT, h = (u / TT) & 3, v = u / (TT * HH);
        const int c0 = h * DD, rowi = i * VV + v;
        float qv[DD];
#pragma unroll
        for (int d = 0; d < DD; ++d) qv[d] = (float)sQ[rowi * EE + c0 + d];
        float sc[TT], mx = -1e30f;
#pragma unroll
        for (int j = 0; j < TT; ++j) {
            const int rj = j * VV + v;
            float s = 0.f;
#pragma unroll
            for (int d = 0; d < DD; ++d) s += qv[d] * (float)sK[rj * EE + c0 + d];
            s *= 0.25f; sc[j] = s; mx = fmaxf(mx, s);
        }
        float den = 0.f;
#pragma unroll
        for (int j = 0; j < TT; ++j) { sc[j] = expf(sc[j] - mx); den += sc[j]; }
        const float inv = 1.f / den;
        float ov[DD];
#pragma unroll
        for (int d = 0; d < DD; ++d) ov[d] = 0.f;
#pragma unroll
        for (int j = 0; j < TT; ++j) {
            const float pw = sc[j] * inv; const int rj = j * VV + v;
#pragma unroll
            for (int d = 0; d < DD; ++d) ov[d] += pw * (float)sV[rj * EE + c0 + d];
        }
#pragma unroll
        for (int d = 0; d < DD; ++d) sQ[rowi * EE + c0 + d] = (_Float16)ov[d];
    }
    __syncthreads();
    stageW(p.ta_wo, p.ta_bo, sWt, sBias, tid); __syncthreads();
    gemmEE(sQ, sX, sWt, sBias, sX, false, lane, wave); __syncthreads();

    // ---------------- stage 6: global max pool + FC ------------------------
    if (tid < EE) {
        float m = -1e30f;
        for (int tok = 0; tok < TOK; ++tok)
            m = fmaxf(m, (float)sX[tok * EE + tid]);
        sPool[tid] = m;
    }
    __syncthreads();
    if (tid < NC) {
        float a = p.fc_b[tid];
        for (int e = 0; e < EE; ++e) a += sPool[e] * p.fc_w[e * NC + tid];
        p.out[(size_t)n * NC + tid] = a;
    }
}

// ---------------------------------------------------------------------------
extern "C" void kernel_launch(void* const* d_in, const int* in_sizes, int n_in,
                              void* d_out, int out_size, void* d_ws, size_t ws_size,
                              hipStream_t stream) {
    Params p;
    p.x_joint  = (const float*)d_in[0];  p.x_vel   = (const float*)d_in[1];
    p.je_w     = (const float*)d_in[2];  p.je_b    = (const float*)d_in[3];
    p.ve_w     = (const float*)d_in[4];  p.ve_b    = (const float*)d_in[5];
    p.joint_emb= (const float*)d_in[6];  p.frame_emb=(const float*)d_in[7];
    p.proj_w   = (const float*)d_in[8];  p.proj_b  = (const float*)d_in[9];
    p.gcn_w1   = (const float*)d_in[10]; p.gcn_b1  = (const float*)d_in[11];
    p.gcn_w2   = (const float*)d_in[12]; p.gcn_b2  = (const float*)d_in[13];
    p.sa_wq = (const float*)d_in[14]; p.sa_bq = (const float*)d_in[15];
    p.sa_wk = (const float*)d_in[16]; p.sa_bk = (const float*)d_in[17];
    p.sa_wv = (const float*)d_in[18]; p.sa_bv = (const float*)d_in[19];
    p.sa_wo = (const float*)d_in[20]; p.sa_bo = (const float*)d_in[21];
    p.ta_wq = (const float*)d_in[22]; p.ta_bq = (const float*)d_in[23];
    p.ta_wk = (const float*)d_in[24]; p.ta_bk = (const float*)d_in[25];
    p.ta_wv = (const float*)d_in[26]; p.ta_bv = (const float*)d_in[27];
    p.ta_wo = (const float*)d_in[28]; p.ta_bo = (const float*)d_in[29];
    p.fc_w  = (const float*)d_in[30]; p.fc_b  = (const float*)d_in[31];
    p.terms = (const float*)d_ws;
    p.out   = (float*)d_out;
    p.n_batch = in_sizes[0] / (CCH * TT * VV);

    precompute_terms<<<1, 256, 0, stream>>>(p.joint_emb, p.frame_emb,
                                            p.proj_w, (float*)d_ws);
    st_sagcn_fused<<<p.n_batch, 256, 0, stream>>>(p);
}
